// GPT_7988639170738
// MI455X (gfx1250) — matrix operand order repaired and checked by reading, not compiled, not run on hardware
//
#include <hip/hip_runtime.h>
#include <stdint.h>

typedef __attribute__((ext_vector_type(16))) _Float16 v16h;
typedef __attribute__((ext_vector_type(8)))  float    v8f;

#define N_HEAD 8
#define HD 128
#define FLAG_GELU 1
#define FLAG_RES  2

#ifndef __has_builtin
#define __has_builtin(x) 0
#endif
#if __has_builtin(__builtin_amdgcn_global_load_async_to_lds_b128) && \
    __has_builtin(__builtin_amdgcn_s_wait_asynccnt)
#define HAVE_ASYNC_LDS 1
#else
#define HAVE_ASYNC_LDS 0
#endif

#if HAVE_ASYNC_LDS
typedef int v4i __attribute__((vector_size(16)));
typedef __attribute__((address_space(1))) v4i* gv4i_p;   // global int4*
typedef __attribute__((address_space(3))) v4i* lv4i_p;   // LDS int4*
// 16-byte global -> LDS async copy (VGLOBAL async, tracked by ASYNCcnt)
__device__ __forceinline__ void async_cp16(const void* g, void* l) {
    __builtin_amdgcn_global_load_async_to_lds_b128((gv4i_p)g, (lv4i_p)l, 0, 0);
}
__device__ __forceinline__ void wait_async0() {
    __builtin_amdgcn_s_wait_asynccnt(0);
}
#endif

// ---------------------------------------------------------------------------
// WMMA fragment loaders (wave32, 16x16x32 f16 shape)
// A-matrix 16x32 f16: lane = m(0..15) + 16*khalf, element e holds K = khalf*16+e
//   (same k order as the B fragment)
__device__ __forceinline__ v16h load_a_frag(const _Float16* base, int row, int ldh, int lane) {
    int khalf = lane >> 4;
    return *(const v16h*)(base + (size_t)row * ldh + khalf * 16);
}

// B-matrix 32x16 f16: lane = n(0..15) + 16*khalf, element e holds K = khalf*16+e.
// With B stored [n][k] (k contiguous), a fragment is one 32B contiguous LDS load.
__device__ __forceinline__ v16h load_b_frag(const _Float16* base, int n, int ldh, int lane) {
    int khalf = lane >> 4;
    return *(const v16h*)(base + (size_t)n * ldh + khalf * 16);
}

__device__ __forceinline__ v8f wmma_f16(v16h a, v16h b, v8f c) {
    return __builtin_amdgcn_wmma_f32_16x16x32_f16(false, a, false, b, (short)0, c, false, false);
}

// ---------------------------------------------------------------------------
// h = x + pos_emb (broadcast over batch)
__global__ void k_addpos(float* __restrict__ h, const float* __restrict__ x,
                         const float* __restrict__ pos, int T, int C, size_t total) {
    size_t i = (size_t)blockIdx.x * blockDim.x + threadIdx.x;
    if (i < total) {
        size_t tc = i % ((size_t)T * C);
        h[i] = x[i] + pos[tc];
    }
}

// ---------------------------------------------------------------------------
// LayerNorm over C=1024, one row per block (256 threads), f16 output
__global__ void k_ln(_Float16* __restrict__ out, const float* __restrict__ in,
                     const float* __restrict__ g, const float* __restrict__ b, int C) {
    int row = blockIdx.x, tid = threadIdx.x;
    const float* x = in + (size_t)row * C;
    float v0[4], s = 0.f;
#pragma unroll
    for (int i = 0; i < 4; ++i) { v0[i] = x[tid + 256 * i]; s += v0[i]; }
    __shared__ float red[256];
    red[tid] = s; __syncthreads();
    for (int o = 128; o > 0; o >>= 1) { if (tid < o) red[tid] += red[tid + o]; __syncthreads(); }
    float mean = red[0] / (float)C;
    __syncthreads();
    float s2 = 0.f;
#pragma unroll
    for (int i = 0; i < 4; ++i) { float d = v0[i] - mean; s2 += d * d; }
    red[tid] = s2; __syncthreads();
    for (int o = 128; o > 0; o >>= 1) { if (tid < o) red[tid] += red[tid + o]; __syncthreads(); }
    float inv = rsqrtf(red[0] / (float)C + 1e-5f);
#pragma unroll
    for (int i = 0; i < 4; ++i) {
        int c = tid + 256 * i;
        out[(size_t)row * C + c] = (_Float16)((v0[i] - mean) * inv * g[c] + b[c]);
    }
}

// ---------------------------------------------------------------------------
// C[M,N] = A_f16[M,K] @ W_f32[K,N] (+bias, opt GELU, opt residual add)
// 256 threads = 8 waves, 128x128 tile, K stepped by 32 through LDS.
// A tile staged with GLOBAL_LOAD_ASYNC_TO_LDS_B128 when available.
__global__ void k_gemm(const _Float16* __restrict__ A, const float* __restrict__ W,
                       const float* __restrict__ bias,
                       float* __restrict__ outF32, _Float16* __restrict__ outF16,
                       const float* __restrict__ resid,
                       int M, int N, int K, int flags) {
    int tid = threadIdx.x, lane = tid & 31, w = tid >> 5;
    int n0 = blockIdx.x * 128, m0 = blockIdx.y * 128;
    __shared__ __attribute__((aligned(32))) _Float16 Asl[128 * 32];
    __shared__ __attribute__((aligned(32))) _Float16 Bsl[128 * 32]; // [n][k]

    v8f zero8 = {0.f, 0.f, 0.f, 0.f, 0.f, 0.f, 0.f, 0.f};
    v8f acc[8];
#pragma unroll
    for (int nt = 0; nt < 8; ++nt) acc[nt] = zero8;

    int nl = lane & 15;
    for (int k0 = 0; k0 < K; k0 += 32) {
        __syncthreads();
#if HAVE_ASYNC_LDS
        // A tile: 128 rows x 64B -> 512 b128 async transfers
        for (int i = tid; i < 512; i += 256) {
            int r = i >> 2, seg = i & 3;
            async_cp16(A + (size_t)(m0 + r) * K + k0 + seg * 8, Asl + r * 32 + seg * 8);
        }
#else
        for (int i = tid; i < 128 * 32; i += 256) {
            int r = i >> 5, c = i & 31;
            Asl[i] = A[(size_t)(m0 + r) * K + k0 + c];
        }
#endif
        // B tile transposed + converted: Bsl[n][k] = (f16) W[k0+k][n0+n]
        for (int i = tid; i < 32 * 128; i += 256) {
            int n = i & 127, kk = i >> 7;
            Bsl[n * 32 + kk] = (_Float16)W[(size_t)(k0 + kk) * N + n0 + n];
        }
#if HAVE_ASYNC_LDS
        wait_async0();
#endif
        __syncthreads();

        v16h af = load_a_frag(Asl, 16 * w + nl, 32, lane);
#pragma unroll
        for (int nt = 0; nt < 8; ++nt) {
            v16h bf = load_b_frag(Bsl, 16 * nt + nl, 32, lane);
            acc[nt] = wmma_f16(af, bf, acc[nt]);
        }
    }

    int mh = lane >> 4;
#pragma unroll
    for (int nt = 0; nt < 8; ++nt) {
#pragma unroll
        for (int r = 0; r < 8; ++r) {
            int row = m0 + 16 * w + 8 * mh + r;
            int col = n0 + 16 * nt + nl;
            float v = acc[nt][r] + bias[col];
            if (flags & FLAG_GELU) v = 0.5f * v * (1.f + erff(v * 0.70710678118f));
            if (flags & FLAG_RES)  v += resid[(size_t)row * N + col];
            if (outF32) outF32[(size_t)row * N + col] = v;
            if (outF16) outF16[(size_t)row * N + col] = (_Float16)v;
        }
    }
}

// ---------------------------------------------------------------------------
// Flash attention: per (b, head, 64-query tile). q/k/v f16 [B*T, C], head h at
// column h*128. Graph bias on first 60x60, causal mask, col%64==63 disabled.
__global__ void k_attn(_Float16* __restrict__ y, const _Float16* __restrict__ q,
                       const _Float16* __restrict__ k, const _Float16* __restrict__ v,
                       const float* __restrict__ attn_bias, int T, int C) {
    int m0 = blockIdx.x * 64;
    int h = blockIdx.y;
    int b = blockIdx.z;
    int tid = threadIdx.x, lane = tid & 31, w = tid >> 5;
    int nl = lane & 15, mh = lane >> 4;
    const float scale = 0.088388347648318447f; // 1/sqrt(128)

    __shared__ __attribute__((aligned(32))) _Float16 Qs[64 * 128];
    __shared__ __attribute__((aligned(32))) _Float16 Ks[64 * 128];
    __shared__ __attribute__((aligned(32))) _Float16 Vs[128 * 64]; // transposed [dim][key]
    __shared__ __attribute__((aligned(32))) _Float16 Ps[64 * 64];

    size_t base = (size_t)b * T * C + (size_t)h * HD;
#if HAVE_ASYNC_LDS
    for (int i = tid; i < 1024; i += 128) {       // 64 rows x 16 b128 segs
        int r = i >> 4, seg = i & 15;
        async_cp16(q + base + (size_t)(m0 + r) * C + seg * 8, Qs + r * 128 + seg * 8);
    }
#else
    for (int i = tid; i < 64 * 128; i += 128) {
        int r = i >> 7, d = i & 127;
        Qs[i] = q[base + (size_t)(m0 + r) * C + d];
    }
#endif

    v8f zero8 = {0.f, 0.f, 0.f, 0.f, 0.f, 0.f, 0.f, 0.f};
    v8f o[8];
#pragma unroll
    for (int t8 = 0; t8 < 8; ++t8) o[t8] = zero8;
    float mrow[8], lrow[8];
#pragma unroll
    for (int r = 0; r < 8; ++r) { mrow[r] = -1e30f; lrow[r] = 0.f; }

    for (int j0 = 0; j0 <= m0; j0 += 64) {
        __syncthreads();
#if HAVE_ASYNC_LDS
        for (int i = tid; i < 1024; i += 128) {
            int r = i >> 4, seg = i & 15;
            async_cp16(k + base + (size_t)(j0 + r) * C + seg * 8, Ks + r * 128 + seg * 8);
        }
        for (int i = tid; i < 64 * 128; i += 128) {
            int r = i >> 7, d = i & 127;
            Vs[d * 64 + r] = v[base + (size_t)(j0 + r) * C + d];
        }
        wait_async0();
#else
        for (int i = tid; i < 64 * 128; i += 128) {
            int r = i >> 7, d = i & 127;
            Ks[i] = k[base + (size_t)(j0 + r) * C + d];
            Vs[d * 64 + r] = v[base + (size_t)(j0 + r) * C + d];
        }
#endif
        __syncthreads();

        v8f s[4];
#pragma unroll
        for (int nt = 0; nt < 4; ++nt) s[nt] = zero8;
#pragma unroll
        for (int kk = 0; kk < 128; kk += 32) {
            v16h af = load_a_frag(Qs + kk, 16 * w + nl, 128, lane);
#pragma unroll
            for (int nt = 0; nt < 4; ++nt) {
                v16h bf = load_b_frag(Ks + kk, 16 * nt + nl, 128, lane);
                s[nt] = wmma_f16(af, bf, s[nt]);
            }
        }
        // scale + graph bias + mask
#pragma unroll
        for (int nt = 0; nt < 4; ++nt) {
#pragma unroll
            for (int r = 0; r < 8; ++r) {
                int qi = m0 + 16 * w + 8 * mh + r;
                int kj = j0 + 16 * nt + nl;
                float val = s[nt][r] * scale;
                if (qi < 60 && kj < 60)
                    val += attn_bias[(h * 15 + (qi >> 2)) * 15 + (kj >> 2)];
                bool ok = (kj <= qi) && ((kj & 63) != 63);
                s[nt][r] = ok ? val : -1e30f;
            }
        }
        // online softmax per row (16-lane reductions within each half-wave)
#pragma unroll
        for (int r = 0; r < 8; ++r) {
            float rmax = fmaxf(fmaxf(s[0][r], s[1][r]), fmaxf(s[2][r], s[3][r]));
#pragma unroll
            for (int off = 8; off > 0; off >>= 1) rmax = fmaxf(rmax, __shfl_xor(rmax, off));
            float mnew = fmaxf(mrow[r], rmax);
            float alpha = __expf(mrow[r] - mnew);
            float rsum = 0.f;
#pragma unroll
            for (int nt = 0; nt < 4; ++nt) {
                float sv = s[nt][r];
                float pv = (sv < -1e29f) ? 0.f : __expf(sv - mnew);
                rsum += pv;
                Ps[(16 * w + 8 * mh + r) * 64 + 16 * nt + nl] = (_Float16)pv;
            }
#pragma unroll
            for (int off = 8; off > 0; off >>= 1) rsum += __shfl_xor(rsum, off);
            lrow[r] = lrow[r] * alpha + rsum;
            mrow[r] = mnew;
#pragma unroll
            for (int t8 = 0; t8 < 8; ++t8) o[t8][r] *= alpha;
        }
        __syncthreads();
        // O += P @ V   (K-dim = 64 keys, two 32-chunks)
#pragma unroll
        for (int kk2 = 0; kk2 < 64; kk2 += 32) {
            v16h af = load_a_frag(Ps + kk2, 16 * w + nl, 64, lane);
#pragma unroll
            for (int nt2 = 0; nt2 < 8; ++nt2) {
                v16h bf = load_b_frag(Vs + kk2, 16 * nt2 + nl, 64, lane);
                o[nt2] = wmma_f16(af, bf, o[nt2]);
            }
        }
    }

#pragma unroll
    for (int r = 0; r < 8; ++r) {
        float inv = lrow[r] > 0.f ? 1.f / lrow[r] : 0.f;
        int qi = m0 + 16 * w + 8 * mh + r;
#pragma unroll
        for (int nt2 = 0; nt2 < 8; ++nt2)
            y[base + (size_t)qi * C + 16 * nt2 + nl] = (_Float16)(o[nt2][r] * inv);
    }
}

// ---------------------------------------------------------------------------
// EinLinear head: logits[b,t,o] = dot(hf[b,t,:], head_w[t%64, o, :])
__global__ void k_head(float* __restrict__ out, const _Float16* __restrict__ hf,
                       const float* __restrict__ head_w, int T, int C, int V1) {
    int token = blockIdx.x;           // b*T + t
    int e = (token % T) & 63;
    __shared__ float hrow[1024];
    for (int i = threadIdx.x; i < C; i += blockDim.x)
        hrow[i] = (float)hf[(size_t)token * C + i];
    __syncthreads();
    for (int oo = threadIdx.x; oo < V1; oo += blockDim.x) {
        const float* wr = head_w + ((size_t)e * V1 + oo) * C;
        float s = 0.f;
        for (int i = 0; i < C; ++i) s += hrow[i] * wr[i];
        out[(size_t)token * V1 + oo] = s;
    }
}

// ---------------------------------------------------------------------------
extern "C" void kernel_launch(void* const* d_in, const int* in_sizes, int n_in,
                              void* d_out, int out_size, void* d_ws, size_t ws_size,
                              hipStream_t stream) {
    (void)in_sizes; (void)n_in; (void)out_size; (void)ws_size;
    const float* x         = (const float*)d_in[0];
    const float* attn_bias = (const float*)d_in[1];
    const float* pos       = (const float*)d_in[2];
    const float* Wq = (const float*)d_in[3];  const float* bq = (const float*)d_in[4];
    const float* Wk = (const float*)d_in[5];  const float* bk = (const float*)d_in[6];
    const float* Wv = (const float*)d_in[7];  const float* bv = (const float*)d_in[8];
    const float* Wp = (const float*)d_in[9];  const float* bp = (const float*)d_in[10];
    const float* ln1g = (const float*)d_in[11]; const float* ln1b = (const float*)d_in[12];
    const float* ln2g = (const float*)d_in[13]; const float* ln2b = (const float*)d_in[14];
    const float* w1 = (const float*)d_in[15]; const float* b1 = (const float*)d_in[16];
    const float* w2 = (const float*)d_in[17]; const float* b2 = (const float*)d_in[18];
    const float* lnfg = (const float*)d_in[19]; const float* lnfb = (const float*)d_in[20];
    const float* head_w = (const float*)d_in[21];

    const int B = 4, T = 1024, C = 1024, L = 4, V1 = 101;
    const size_t BT = (size_t)B * T;

    char* ws = (char*)d_ws;
    float*     h    = (float*)ws;      ws += BT * C * sizeof(float);
    _Float16*  a16  = (_Float16*)ws;   ws += BT * C * sizeof(_Float16);
    _Float16*  q16  = (_Float16*)ws;   ws += BT * C * sizeof(_Float16);
    _Float16*  k16  = (_Float16*)ws;   ws += BT * C * sizeof(_Float16);
    _Float16*  v16b = (_Float16*)ws;   ws += BT * C * sizeof(_Float16);
    _Float16*  y16  = (_Float16*)ws;   ws += BT * C * sizeof(_Float16);
    _Float16*  mid  = (_Float16*)ws;   ws += BT * (size_t)(4 * C) * sizeof(_Float16);

    const size_t total = BT * C;
    k_addpos<<<dim3((unsigned)((total + 255) / 256)), dim3(256), 0, stream>>>(h, x, pos, T, C, total);

    dim3 gSq(C / 128, (unsigned)(BT / 128));        // N=1024 GEMMs
    dim3 gWide(4 * C / 128, (unsigned)(BT / 128));  // N=4096 GEMM
    dim3 gblk(256);
    dim3 ablk(128);

    for (int l = 0; l < L; ++l) {
        k_ln<<<dim3((unsigned)BT), dim3(256), 0, stream>>>(a16, h, ln1g + (size_t)l * C, ln1b + (size_t)l * C, C);
        k_gemm<<<gSq, gblk, 0, stream>>>(a16, Wq + (size_t)l * C * C, bq + (size_t)l * C,
                                         nullptr, q16, nullptr, (int)BT, C, C, 0);
        k_gemm<<<gSq, gblk, 0, stream>>>(a16, Wk + (size_t)l * C * C, bk + (size_t)l * C,
                                         nullptr, k16, nullptr, (int)BT, C, C, 0);
        k_gemm<<<gSq, gblk, 0, stream>>>(a16, Wv + (size_t)l * C * C, bv + (size_t)l * C,
                                         nullptr, v16b, nullptr, (int)BT, C, C, 0);
        k_attn<<<dim3(T / 64, N_HEAD, B), ablk, 0, stream>>>(y16, q16, k16, v16b, attn_bias, T, C);
        k_gemm<<<gSq, gblk, 0, stream>>>(y16, Wp + (size_t)l * C * C, bp + (size_t)l * C,
                                         h, nullptr, h, (int)BT, C, C, FLAG_RES);
        k_ln<<<dim3((unsigned)BT), dim3(256), 0, stream>>>(a16, h, ln2g + (size_t)l * C, ln2b + (size_t)l * C, C);
        k_gemm<<<gWide, gblk, 0, stream>>>(a16, w1 + (size_t)l * C * 4 * C, b1 + (size_t)l * 4 * C,
                                           nullptr, mid, nullptr, (int)BT, 4 * C, C, FLAG_GELU);
        k_gemm<<<gSq, gblk, 0, stream>>>(mid, w2 + (size_t)l * 4 * C * C, b2 + (size_t)l * C,
                                         h, nullptr, h, (int)BT, C, 4 * C, FLAG_RES);
    }

    k_ln<<<dim3((unsigned)BT), dim3(256), 0, stream>>>(a16, h, lnfg, lnfb, C);
    k_head<<<dim3((unsigned)BT), dim3(128), 0, stream>>>((float*)d_out, a16, head_w, T, C, V1);
}
